// NeuroSSMLayer_58695023067777
// MI455X (gfx1250) — matrix operand order repaired and checked by reading, not compiled, not run on hardware
//
#include <hip/hip_runtime.h>
#include <hip/hip_bf16.h>
#include <cstdint>
#include <cstddef>

// ---------------------------------------------------------------------------
// NeuroSSM layer for MI455X (gfx1250): bf16 WMMA everywhere, f32 accumulate.
// Scan uses GLOBAL_LOAD_ASYNC_TO_LDS (ASYNCcnt) for xb double-buffering.
// ---------------------------------------------------------------------------

typedef __attribute__((ext_vector_type(16))) __bf16 v16bf;
typedef __attribute__((ext_vector_type(8)))  float  v8f;

#define T_SEQ 2048
#define BATCH 8
#define HID   1024
#define STATE 256
#define FF    4096
#define NROW  (BATCH * T_SEQ)   // 16384 rows of x / y

// round-to-nearest-even f32 -> bf16
__device__ __forceinline__ unsigned short f2bf(float f) {
  unsigned u = __float_as_uint(f);
  u += 0x7FFFu + ((u >> 16) & 1u);
  return (unsigned short)(u >> 16);
}

union ABf { uint4 u[2]; v16bf v; };

// A operand (16x32 bf16, M x K): per lane, elems 0-7 = K[kc+8*hi .. +7],
// elems 8-15 = K[kc+16+8*hi .. +7]; row m supplied per-lane by caller.
__device__ __forceinline__ v16bf load_a16(const unsigned short* base, int ldk,
                                          int m, int kc, int hi) {
  ABf r;
  const unsigned short* p = base + (size_t)m * ldk + kc + 8 * hi;
  r.u[0] = *(const uint4*)p;
  r.u[1] = *(const uint4*)(p + 16);
  return r.v;
}
// B operand (32x16 bf16, K x N) read from row-major W[N][K]: per lane,
// elems 0-7 = K[kc+8*hi .. +7], elems 8-15 = K[kc+16+8*hi .. +7] of row n.
__device__ __forceinline__ v16bf load_b16(const unsigned short* base, int ldk,
                                          int n, int kc, int hi) {
  ABf r;
  const unsigned short* p = base + (size_t)n * ldk + kc + 8 * hi;
  r.u[0] = *(const uint4*)p;
  r.u[1] = *(const uint4*)(p + 16);
  return r.v;
}

// ---- CDNA5 async global->LDS copy (tracked by ASYNCcnt) -------------------
// VGLOBAL encoding: VDST = LDS byte offset VGPR (LDS aperture is entirely in
// addr[63:32], so the low 32 bits of a generic LDS pointer ARE the offset),
// VADDR = 64-bit global address, saddr = off (null).
__device__ __forceinline__ void async_copy_b128(const float* gsrc,
                                                float* ldst) {
  unsigned loff = (unsigned)(unsigned long long)ldst;
  asm volatile("global_load_async_to_lds_b128 %0, %1, off" ::"v"(loff),
               "v"(gsrc)
               : "memory");
}
__device__ __forceinline__ void async_wait0() {
#if __has_builtin(__builtin_amdgcn_s_wait_asynccnt)
  __builtin_amdgcn_s_wait_asynccnt(0);
#else
  asm volatile("s_wait_asynccnt 0" ::: "memory");
#endif
}

// --------------------------- convert f32 -> bf16 ---------------------------
__global__ void cvt_bf16_kernel(const float* __restrict__ src,
                                unsigned short* __restrict__ dst, int n) {
  int i = blockIdx.x * 256 + threadIdx.x;
  if (i < n) dst[i] = f2bf(src[i]);
}

// --------------------------- generic WMMA GEMM -----------------------------
// out[M,N] = A_bf16[M,K] @ W_bf16[N,K]^T  (+ per-mode epilogue)
// block = 128 threads = 4 waves; each wave computes a 64x64 tile.
#define GM_STORE 0   // out0[m*N+n] = v            (xD)
#define GM_XB    1   // out0[(t*8+b)*256+n] = v    (xB scatter, m = b*2048+t)
#define GM_SSM   2   // states@C^T + xD + x -> y0  (in-place into out0=add0)
#define GM_FFN1  3   // gelu(v + b1) -> bf16 h
#define GM_FFN2  4   // v + b2 + y_ln -> d_out
__global__ void __launch_bounds__(128)
gemm_wmma(const unsigned short* __restrict__ Abf,
          const unsigned short* __restrict__ Wbf,
          int M, int N, int K, int mode,
          float* __restrict__ out0, const float* __restrict__ add0,
          const float* __restrict__ add1, const float* __restrict__ bias,
          unsigned short* __restrict__ outbf) {
  const int lane = threadIdx.x & 31;
  const int wv   = threadIdx.x >> 5;
  const int gw   = blockIdx.x * 4 + wv;
  const int tilesN = N >> 6;
  const int tM = (gw / tilesN) << 6;
  const int tN = (gw % tilesN) << 6;
  const int hi = lane >> 4, l15 = lane & 15;

  v8f acc[4][4] = {};
  for (int kc = 0; kc < K; kc += 32) {
    v16bf a[4], b[4];
#pragma unroll
    for (int i = 0; i < 4; ++i)
      a[i] = load_a16(Abf, K, tM + 16 * i + l15, kc, hi);
#pragma unroll
    for (int j = 0; j < 4; ++j)
      b[j] = load_b16(Wbf, K, tN + 16 * j + l15, kc, hi);
    if (kc + 32 < K) {  // hint next K-slice into cache (global_prefetch_b8)
      __builtin_prefetch(Abf + (size_t)(tM + l15) * K + kc + 32, 0, 1);
      __builtin_prefetch(Wbf + (size_t)(tN + l15) * K + kc + 32, 0, 1);
    }
#pragma unroll
    for (int i = 0; i < 4; ++i)
#pragma unroll
      for (int j = 0; j < 4; ++j)
        acc[i][j] = __builtin_amdgcn_wmma_f32_16x16x32_bf16(
            false, a[i], false, b[j], (short)0, acc[i][j], false, false);
  }

#pragma unroll
  for (int i = 0; i < 4; ++i)
#pragma unroll
    for (int j = 0; j < 4; ++j)
#pragma unroll
      for (int r = 0; r < 8; ++r) {
        const int m = tM + 16 * i + r + 8 * hi;
        const int n = tN + 16 * j + l15;
        float v = acc[i][j][r];
        if (mode == GM_STORE) {
          out0[(size_t)m * N + n] = v;
        } else if (mode == GM_XB) {
          const int b = m >> 11, t = m & (T_SEQ - 1);
          out0[((size_t)t * BATCH + b) * STATE + n] = v;
        } else if (mode == GM_SSM) {
          const int t = m >> 4, mm = m & 15;
          if (mm < 8) {
            const size_t idx = ((size_t)mm * T_SEQ + t) * HID + n;
            out0[idx] = v + add0[idx] + add1[idx];
          }
        } else if (mode == GM_FFN1) {
          v += bias[n];
          v = 0.5f * v * (1.0f + erff(v * 0.70710678118654752f));  // exact GELU
          outbf[(size_t)m * N + n] = f2bf(v);
        } else {  // GM_FFN2
          const size_t idx = (size_t)m * N + n;
          out0[idx] = v + bias[n] + add0[idx];
        }
      }
}

// ------------------------------- SSM scan ----------------------------------
// One workgroup, 512 threads = 16 waves. State lives in LDS as a 16x256 bf16
// WMMA A-tile (rows 0-7 = batch, rows 8-15 identically zero: GEMM rows are
// independent and tanh(0*A+0)=0). Wave w owns output columns [16w,16w+16),
// keeping its 8 A^T B-operand chunks in VGPRs for the whole 2048-step loop.
// xb[t+1] is prefetched with GLOBAL_LOAD_ASYNC_TO_LDS into a double buffer.
#define STS 264   // LDS row stride (bf16 elems): 528B = 33*16B, kills conflicts
__global__ void __launch_bounds__(512)
scan_kernel(const unsigned short* __restrict__ Abf /*[256][256]*/,
            const float* __restrict__ xB /*[T][8][256] f32*/,
            unsigned short* __restrict__ states /*[T][16][256] bf16*/,
            float* __restrict__ fs_out /*[8][256] f32*/) {
  __shared__ unsigned short st[16 * STS];
  __shared__ float xbs[2][BATCH * STATE];  // double-buffered xb tile
  const int tid = threadIdx.x;
  const int lane = tid & 31, wv = tid >> 5;
  const int hi = lane >> 4, l15 = lane & 15;
  const int nc = wv * 16 + l15;  // this lane's output column

  for (int i = tid; i < 16 * STS; i += 512) st[i] = 0;

  // Preload B operands: elem (k,n) = A[n][k]  (state @ A^T)
  v16bf bop[8];
#pragma unroll
  for (int j = 0; j < 8; ++j) {
    ABf r;
    const unsigned short* p = Abf + (size_t)nc * STATE + 32 * j + 8 * hi;
    r.u[0] = *(const uint4*)p;
    r.u[1] = *(const uint4*)(p + 16);
    bop[j] = r.v;
  }
  // async-load xb[0] into buffer 0 (one b128 per thread covers 8x256 f32)
  async_copy_b128(xB + tid * 4, &xbs[0][tid * 4]);
  async_wait0();
  __syncthreads();

  for (int t = 0; t < T_SEQ; ++t) {
    const int cur = t & 1;
    if (t + 1 < T_SEQ)  // async prefetch next xb while WMMAs run
      async_copy_b128(xB + (size_t)(t + 1) * (BATCH * STATE) + tid * 4,
                      &xbs[cur ^ 1][tid * 4]);

    // A operands from LDS state
    v16bf aop[8];
#pragma unroll
    for (int j = 0; j < 8; ++j) {
      ABf r;
      const unsigned short* p = st + l15 * STS + 32 * j + 8 * hi;
      r.u[0] = *(const uint4*)p;
      r.u[1] = *(const uint4*)(p + 16);
      aop[j] = r.v;
    }

    v8f acc = {};
#pragma unroll
    for (int j = 0; j < 8; ++j)
      acc = __builtin_amdgcn_wmma_f32_16x16x32_bf16(
          false, aop[j], false, bop[j], (short)0, acc, false, false);

    float nv[8];
#pragma unroll
    for (int r = 0; r < 8; ++r) {
      float v = acc[r];
      if (!hi) v += xbs[cur][r * STATE + nc];  // rows 8-15 get +0
      nv[r] = tanhf(v);
    }
    __syncthreads();  // all reads of st / xbs[cur] complete

    unsigned short* sg = states + (size_t)t * (16 * STATE);
#pragma unroll
    for (int r = 0; r < 8; ++r) {
      const int m = r + 8 * hi;
      const unsigned short bb = f2bf(nv[r]);
      st[m * STS + nc] = bb;
      sg[m * STATE + nc] = bb;
    }
    if (t == T_SEQ - 1 && !hi) {
#pragma unroll
      for (int r = 0; r < 8; ++r) fs_out[r * STATE + nc] = nv[r];
    }
    async_wait0();    // this wave's async chunk of xb[t+1] is in LDS
    __syncthreads();  // ... and everyone else's too; st fully updated
  }
}

// ------------------------------- LayerNorm ---------------------------------
__global__ void __launch_bounds__(256)
ln_kernel(float* __restrict__ y /*in-place [NROW][HID]*/,
          unsigned short* __restrict__ ybf,
          const float* __restrict__ w, const float* __restrict__ b) {
  __shared__ float red[18];
  const int row = blockIdx.x;
  float* p = y + (size_t)row * HID;
  float v[4], s = 0.f, s2 = 0.f;
#pragma unroll
  for (int i = 0; i < 4; ++i) {
    v[i] = p[threadIdx.x * 4 + i];
    s += v[i];
    s2 += v[i] * v[i];
  }
#pragma unroll
  for (int o = 16; o; o >>= 1) {
    s  += __shfl_down(s, o, 32);
    s2 += __shfl_down(s2, o, 32);
  }
  const int wv = threadIdx.x >> 5, lane = threadIdx.x & 31;
  __shared__ float part[16];
  if (lane == 0) { part[wv] = s; part[8 + wv] = s2; }
  __syncthreads();
  if (threadIdx.x == 0) {
    float ts = 0.f, ts2 = 0.f;
    for (int i = 0; i < 8; ++i) { ts += part[i]; ts2 += part[8 + i]; }
    red[16] = ts; red[17] = ts2;
  }
  __syncthreads();
  const float mu  = red[16] * (1.0f / HID);
  const float var = red[17] * (1.0f / HID) - mu * mu;
  const float inv = rsqrtf(var + 1e-5f);
#pragma unroll
  for (int i = 0; i < 4; ++i) {
    const int c = threadIdx.x * 4 + i;
    const float o = (v[i] - mu) * inv * w[c] + b[c];
    p[c] = o;
    ybf[(size_t)row * HID + c] = f2bf(o);
  }
}

// ------------------------------- launcher ----------------------------------
extern "C" void kernel_launch(void* const* d_in, const int* in_sizes, int n_in,
                              void* d_out, int out_size, void* d_ws,
                              size_t ws_size, hipStream_t stream) {
  (void)in_sizes; (void)n_in; (void)out_size; (void)ws_size;
  const float* x   = (const float*)d_in[0];   // [8,2048,1024]
  const float* A   = (const float*)d_in[1];   // [256,256]
  const float* Bm  = (const float*)d_in[2];   // [256,1024]
  const float* C   = (const float*)d_in[3];   // [1024,256]
  const float* D   = (const float*)d_in[4];   // [1024,1024]
  const float* lnw = (const float*)d_in[5];
  const float* lnb = (const float*)d_in[6];
  const float* W1  = (const float*)d_in[7];   // [4096,1024]
  const float* b1  = (const float*)d_in[8];
  const float* W2  = (const float*)d_in[9];   // [1024,4096]
  const float* b2  = (const float*)d_in[10];
  float* out = (float*)d_out;                  // y [8,2048,1024] ++ fs [8,256]

  char* ws = (char*)d_ws;
  size_t off = 0;
  auto alloc = [&](size_t bytes) -> void* {
    void* p = ws + off;
    off = (off + bytes + 255) & ~(size_t)255;
    return p;
  };
  unsigned short* xbf   = (unsigned short*)alloc((size_t)NROW * HID * 2);
  unsigned short* Abf   = (unsigned short*)alloc((size_t)STATE * STATE * 2);
  unsigned short* Bmbf  = (unsigned short*)alloc((size_t)STATE * HID * 2);
  unsigned short* Cbf   = (unsigned short*)alloc((size_t)HID * STATE * 2);
  unsigned short* Dbf   = (unsigned short*)alloc((size_t)HID * HID * 2);
  unsigned short* W1bf  = (unsigned short*)alloc((size_t)FF * HID * 2);
  unsigned short* W2bf  = (unsigned short*)alloc((size_t)HID * FF * 2);
  float* ybuf           = (float*)alloc((size_t)NROW * HID * 4);  // xD->y0->y_ln
  float* xBb            = (float*)alloc((size_t)T_SEQ * BATCH * STATE * 4);
  unsigned short* stbuf = (unsigned short*)alloc((size_t)T_SEQ * 16 * STATE * 2);
  unsigned short* ylnbf = (unsigned short*)alloc((size_t)NROW * HID * 2);
  unsigned short* hbf   = (unsigned short*)alloc((size_t)NROW * FF * 2);

  auto cvt = [&](const float* s, unsigned short* d, int n) {
    cvt_bf16_kernel<<<(n + 255) / 256, 256, 0, stream>>>(s, d, n);
  };
  cvt(x,  xbf,  NROW * HID);
  cvt(A,  Abf,  STATE * STATE);
  cvt(Bm, Bmbf, STATE * HID);
  cvt(C,  Cbf,  HID * STATE);
  cvt(D,  Dbf,  HID * HID);
  cvt(W1, W1bf, FF * HID);
  cvt(W2, W2bf, HID * FF);

  auto gemm = [&](const unsigned short* Ab, const unsigned short* Wb, int M,
                  int N, int K, int mode, float* o0, const float* a0,
                  const float* a1, const float* bi, unsigned short* ob) {
    int blocks = ((M / 64) * (N / 64)) / 4;
    gemm_wmma<<<blocks, 128, 0, stream>>>(Ab, Wb, M, N, K, mode, o0, a0, a1,
                                          bi, ob);
  };

  // xD = x @ D^T                        -> ybuf
  gemm(xbf, Dbf, NROW, HID, HID, GM_STORE, ybuf, nullptr, nullptr, nullptr, nullptr);
  // xB = x @ Bm^T  (scatter to [t][b][s])
  gemm(xbf, Bmbf, NROW, STATE, HID, GM_XB, xBb, nullptr, nullptr, nullptr, nullptr);
  // sequential scan: states[t] = tanh(states[t-1] @ A^T + xB[t])
  scan_kernel<<<1, 512, 0, stream>>>(Abf, xBb, stbuf,
                                     out + (size_t)NROW * HID);
  // y0 = states @ C^T + xD + x          (in-place into ybuf)
  gemm(stbuf, Cbf, T_SEQ * 16, HID, STATE, GM_SSM, ybuf, ybuf, x, nullptr, nullptr);
  // y_ln = LN(y0)                       (in-place f32 + bf16 copy)
  ln_kernel<<<NROW, 256, 0, stream>>>(ybuf, ylnbf, lnw, lnb);
  // h = gelu(y_ln @ W1^T + b1)          -> bf16
  gemm(ylnbf, W1bf, NROW, FF, HID, GM_FFN1, nullptr, nullptr, nullptr, b1, hbf);
  // y = y_ln + h @ W2^T + b2            -> d_out
  gemm(hbf, W2bf, NROW, HID, FF, GM_FFN2, out, ybuf, nullptr, b2, nullptr);
}
